// PromptorL2P_68410239091083
// MI455X (gfx1250) — matrix operand order repaired and checked by reading, compiled, not run on hardware
//
#include <hip/hip_runtime.h>
#include <cstdint>

// ---------------- problem constants ----------------
#define BATCH      16384
#define N_CLOUD    256
#define QD         2048
#define NP         1024     // num prompts
#define PD         1024     // prompt dim
#define HIDDEN     64
#define OUT_DIM    64
#define NSEL       32

typedef __attribute__((ext_vector_type(16))) __bf16 v16bf;
typedef __attribute__((ext_vector_type(8)))  float  v8f;
typedef __attribute__((ext_vector_type(4)))  unsigned int u32x4;
typedef __attribute__((ext_vector_type(8)))  int          i32x8;
typedef __attribute__((ext_vector_type(4)))  int          i32x4;

// ---- CDNA5 Tensor Data Mover (probe-verified builtin; by-value descriptor groups) ----
#if defined(__gfx1250__) && __has_builtin(__builtin_amdgcn_tensor_load_to_lds) && __has_builtin(__builtin_amdgcn_s_wait_tensorcnt)
#define HAS_TDM 1
#else
#define HAS_TDM 0
#endif

#if HAS_TDM
// 2D tile load: rows x 32 dwords (128B) from global (row stride 1024 dwords)
// into LDS with row pitch 144B (pad 4 dwords after every 32 stored dwords).
__device__ __forceinline__ void tdm_tile_load(const void* g, unsigned lds_byte_off, int rows) {
  unsigned long long ga = (unsigned long long)(size_t)g;
  // D# group0: count=1 | lds_addr | global_addr[56:0] | type=2
  u32x4 g0 = { 1u,
               lds_byte_off,
               (unsigned)(ga & 0xffffffffu),
               (unsigned)((ga >> 32) & 0x01ffffffu) | (2u << 30) };
  // D# group1:
  //  d0: wg_mask=0 | data_size=2 (4B) | pad_enable | pad_interval=4 (32 dw) | pad_amount=3 (4 dw)
  //  d1: atomic_barrier_addr=0 | tensor_dim0[15:0]=1024
  //  d2: tensor_dim0[31:16]=0 | tensor_dim1[15:0]=rows
  //  d3: tensor_dim1[31:16]=0 | tile_dim0=32
  //  d4: tile_dim1=rows | tile_dim2=0
  //  d5: tensor_dim0_stride[31:0]=1024  d6,d7: stride hi / dim1_stride = 0
  i32x8 g1 = { (2 << 16) | (1 << 20) | (4 << 22) | (3 << 25),
               (int)(1024u << 16),
               (int)((unsigned)rows << 16),
               (int)(32u << 16),
               rows,
               1024,
               0, 0 };
  i32x4 g2 = { 0, 0, 0, 0 };
  i32x4 g3 = { 0, 0, 0, 0 };
#if __clang_major__ >= 23
  i32x8 g2b = { 0, 0, 0, 0, 0, 0, 0, 0 };
  __builtin_amdgcn_tensor_load_to_lds(g0, g1, g2, g3, g2b, 0);
#else
  __builtin_amdgcn_tensor_load_to_lds(g0, g1, g2, g3, 0);
#endif
}
#endif

// =====================================================================
// Kernel 1: L2-normalize rows of [rows, 2048] f32, emit bf16 rows.
// =====================================================================
__global__ __launch_bounds__(256)
void rownorm_bf16_kernel(const float* __restrict__ in, unsigned short* __restrict__ out)
{
  const int b = blockIdx.x;
  const int t = threadIdx.x;
  const float4* inr = (const float4*)(in + (size_t)b * QD);
  float4 v0 = inr[t * 2 + 0];
  float4 v1 = inr[t * 2 + 1];
  float ss = v0.x*v0.x + v0.y*v0.y + v0.z*v0.z + v0.w*v0.w
           + v1.x*v1.x + v1.y*v1.y + v1.z*v1.z + v1.w*v1.w;
  for (int off = 16; off > 0; off >>= 1) ss += __shfl_xor(ss, off, 32);
  __shared__ float wsum[8];
  __shared__ float sscale;
  if ((t & 31) == 0) wsum[t >> 5] = ss;
  __syncthreads();
  if (t == 0) {
    float s = 0.f;
    for (int i = 0; i < 8; ++i) s += wsum[i];
    sscale = 1.0f / fmaxf(sqrtf(s), 1e-12f);
  }
  __syncthreads();
  const float sc = sscale;
  union { __bf16 h[8]; uint4 u; } o;
  o.h[0] = (__bf16)(v0.x * sc); o.h[1] = (__bf16)(v0.y * sc);
  o.h[2] = (__bf16)(v0.z * sc); o.h[3] = (__bf16)(v0.w * sc);
  o.h[4] = (__bf16)(v1.x * sc); o.h[5] = (__bf16)(v1.y * sc);
  o.h[6] = (__bf16)(v1.z * sc); o.h[7] = (__bf16)(v1.w * sc);
  *(uint4*)(out + (size_t)b * QD + t * 8) = o.u;
}

// =====================================================================
// Kernel 2: dist[B,1024] = 0.5*(1 - q_hat . k_hat) via bf16 WMMA.
// Block tile 64(M) x 256(N), 8 waves in 2x4 grid, each wave 32x64.
// Tiles staged through LDS by the Tensor Data Mover.
// =====================================================================
union ABfrag { v16bf f; uint4 u[2]; };

__global__ __launch_bounds__(256)
void gemm_dist_kernel(const unsigned short* __restrict__ qn_,
                      const unsigned short* __restrict__ kn_,
                      float* __restrict__ dist)
{
  const __bf16* qn = (const __bf16*)qn_;
  const __bf16* kn = (const __bf16*)kn_;

  __shared__ __align__(16) __bf16 As[64][72];    // pitch 144B = 36 dwords
  __shared__ __align__(16) __bf16 Bs[256][72];

  const int t    = threadIdx.x;
  const int lane = t & 31;
  const int warp = t >> 5;
  const int wm   = warp >> 2;     // 0..1
  const int wn   = warp & 3;      // 0..3
  const int bn0  = blockIdx.x * 256;
  const int bm0  = blockIdx.y * 64;
  const int lrow = lane & 15;
  const int hi   = lane >> 4;     // wave half

  v8f acc[2][4];
  for (int i = 0; i < 2; ++i)
    for (int j = 0; j < 4; ++j)
      for (int r = 0; r < 8; ++r) acc[i][j][r] = 0.f;

#if HAS_TDM
  const unsigned As_off = (unsigned)(size_t)&As[0][0];   // flat low 32b = LDS offset
  const unsigned Bs_off = (unsigned)(size_t)&Bs[0][0];
#endif

  for (int kb = 0; kb < QD; kb += 64) {
#if HAS_TDM
    if (warp == 0) {
      tdm_tile_load(&qn[(size_t)bm0 * QD + kb], As_off, 64);
      tdm_tile_load(&kn[(size_t)bn0 * QD + kb], Bs_off, 256);
      __builtin_amdgcn_s_wait_tensorcnt(0);
    }
    __syncthreads();
#else
    for (int i = 0; i < 2; ++i) {
      int c = t + i * 256;
      int row = c >> 3, col = (c & 7) * 8;
      *(uint4*)&As[row][col] = *(const uint4*)&qn[(size_t)(bm0 + row) * QD + kb + col];
    }
    for (int i = 0; i < 8; ++i) {
      int c = t + i * 256;
      int row = c >> 3, col = (c & 7) * 8;
      *(uint4*)&Bs[row][col] = *(const uint4*)&kn[(size_t)(bn0 + row) * QD + kb + col];
    }
    __syncthreads();
#endif

    for (int ks = 0; ks < 64; ks += 32) {
      ABfrag a[2], b[4];
      // A frag (16x32 bf16): half 0 -> K {0..7,16..23}, half 1 -> {8..15,24..31}
      for (int mi = 0; mi < 2; ++mi) {
        const __bf16* src = &As[wm * 32 + mi * 16 + lrow][ks + hi * 8];
        a[mi].u[0] = *(const uint4*)(src);
        a[mi].u[1] = *(const uint4*)(src + 16);
      }
      // B frag (32x16 bf16): half 0 -> K {0..7,16..23}, half 1 -> {8..15,24..31}
      for (int ni = 0; ni < 4; ++ni) {
        const __bf16* src = &Bs[wn * 64 + ni * 16 + lrow][ks + hi * 8];
        b[ni].u[0] = *(const uint4*)(src);
        b[ni].u[1] = *(const uint4*)(src + 16);
      }
      for (int mi = 0; mi < 2; ++mi)
        for (int ni = 0; ni < 4; ++ni)
          acc[mi][ni] = __builtin_amdgcn_wmma_f32_16x16x32_bf16(
              false, a[mi].f, false, b[ni].f, (short)0, acc[mi][ni], false, false);
    }
    __syncthreads();
  }

  // C/D layout: lanes 0-15: N=lane, M=r ; lanes 16-31: N=lane-16, M=r+8
  for (int mi = 0; mi < 2; ++mi)
    for (int ni = 0; ni < 4; ++ni) {
      const int n = bn0 + wn * 64 + ni * 16 + lrow;
      for (int r = 0; r < 8; ++r) {
        const int m = bm0 + wm * 32 + mi * 16 + r + hi * 8;
        dist[(size_t)m * NP + n] = 0.5f * (1.0f - acc[mi][ni][r]);
      }
    }
}

// =====================================================================
// Kernel 3: per-row 32-smallest of 1024 dists (ascending), softmax
// weights; selected_dist -> d_out tail, weights+indices -> ws.
// =====================================================================
__global__ __launch_bounds__(256)
void topk_kernel(const float* __restrict__ dist,
                 float* __restrict__ selw, int* __restrict__ seli,
                 float* __restrict__ outsel)
{
  const int b = blockIdx.x;
  const int t = threadIdx.x;
  __shared__ float sv[NP];
  __shared__ unsigned long long warr[8];
  __shared__ float seld_s[NSEL];
  __shared__ int   seli_s[NSEL];

  for (int j = 0; j < NP / 256; ++j)
    sv[t + j * 256] = dist[(size_t)b * NP + t + j * 256];
  __syncthreads();

  for (int i = 0; i < NSEL; ++i) {
    unsigned long long local = ~0ull;
    for (int j = 0; j < NP / 256; ++j) {
      int n = t + j * 256;
      unsigned u = __float_as_uint(sv[n]);
      u = (u & 0x80000000u) ? ~u : (u | 0x80000000u);   // order-preserving key
      unsigned long long k = ((unsigned long long)u << 32) | (unsigned)n;
      local = (k < local) ? k : local;
    }
    for (int off = 16; off > 0; off >>= 1) {
      unsigned long long o = __shfl_xor(local, off, 32);
      local = (o < local) ? o : local;
    }
    if ((t & 31) == 0) warr[t >> 5] = local;
    __syncthreads();
    if (t == 0) {
      unsigned long long bk = warr[0];
      for (int w = 1; w < 8; ++w) bk = (warr[w] < bk) ? warr[w] : bk;
      int idx = (int)(bk & 0xffffffffu);
      unsigned u = (unsigned)(bk >> 32);
      u = (u & 0x80000000u) ? (u & 0x7fffffffu) : ~u;   // undo key transform
      seld_s[i] = __uint_as_float(u);
      seli_s[i] = idx;
      sv[idx] = __uint_as_float(0x7f800000u);           // +inf: remove from pool
    }
    __syncthreads();
  }

  if (t < NSEL) {
    float d = seld_s[t];
    float s = 1.0f - d;
    float m = s;
    for (int off = 16; off > 0; off >>= 1) m = fmaxf(m, __shfl_xor(m, off, 32));
    float e = __expf(s - m);
    float sum = e;
    for (int off = 16; off > 0; off >>= 1) sum += __shfl_xor(sum, off, 32);
    selw[(size_t)b * NSEL + t] = e / (sum * (float)NSEL);  // fold mean's 1/32
    seli[(size_t)b * NSEL + t] = seli_s[t];
    outsel[(size_t)b * NSEL + t] = d;
  }
}

// =====================================================================
// Kernel 4: gather weighted prompts -> MLP(1024->64 relu ->64) ->
// broadcast over 256 cloud points. One block per batch row.
// =====================================================================
__global__ __launch_bounds__(256)
void gather_mlp_kernel(const float* __restrict__ prompts,
                       const float* __restrict__ selw, const int* __restrict__ seli,
                       const float* __restrict__ w1, const float* __restrict__ b1,
                       const float* __restrict__ w2, const float* __restrict__ b2,
                       float* __restrict__ out)
{
  const int b = blockIdx.x;
  const int t = threadIdx.x;
  __shared__ int   sidx[NSEL];
  __shared__ float sw[NSEL];
  __shared__ __align__(16) float avgs[PD];
  __shared__ float hs[HIDDEN];
  __shared__ __align__(16) float os[OUT_DIM];

  if (t < NSEL) {
    sidx[t] = seli[(size_t)b * NSEL + t];
    sw[t]   = selw[(size_t)b * NSEL + t];
  }
  __syncthreads();

  float4 acc = make_float4(0.f, 0.f, 0.f, 0.f);
  for (int i = 0; i < NSEL; ++i) {
    const float4* pr = (const float4*)(prompts + (size_t)sidx[i] * PD);
    float4 p = pr[t];
    float  w = sw[i];
    acc.x += w * p.x; acc.y += w * p.y; acc.z += w * p.z; acc.w += w * p.w;
  }
  ((float4*)avgs)[t] = acc;
  __syncthreads();

  {
    const int j = t >> 2, part = t & 3;
    const float4* w1r = (const float4*)(w1 + (size_t)j * PD + part * 256);
    const float4* av  = (const float4*)(avgs + part * 256);
    float s = 0.f;
    for (int q = 0; q < 64; ++q) {
      float4 a = av[q], wv = w1r[q];
      s += a.x*wv.x + a.y*wv.y + a.z*wv.z + a.w*wv.w;
    }
    s += __shfl_xor(s, 1, 32);
    s += __shfl_xor(s, 2, 32);
    if (part == 0) hs[j] = fmaxf(s + b1[j], 0.0f);
  }
  __syncthreads();

  if (t < OUT_DIM) {
    float o = b2[t];
    const float* w2r = w2 + (size_t)t * HIDDEN;
    for (int i = 0; i < HIDDEN; ++i) o += hs[i] * w2r[i];
    os[t] = o;
  }
  __syncthreads();

  float4* xrow = (float4*)(out + (size_t)b * (N_CLOUD * OUT_DIM));
  const float4* ov = (const float4*)os;
  for (int it = 0; it < 16; ++it) {
    int c = t + it * 256;
    xrow[c] = ov[c & 15];
  }
}

// =====================================================================
// Launch
// =====================================================================
extern "C" void kernel_launch(void* const* d_in, const int* in_sizes, int n_in,
                              void* d_out, int out_size, void* d_ws, size_t ws_size,
                              hipStream_t stream) {
  const float* query   = (const float*)d_in[1];
  const float* keys    = (const float*)d_in[2];
  const float* prompts = (const float*)d_in[3];
  const float* w1      = (const float*)d_in[4];
  const float* b1      = (const float*)d_in[5];
  const float* w2      = (const float*)d_in[6];
  const float* w2b     = (const float*)d_in[7];
  float* out = (float*)d_out;

  char* ws = (char*)d_ws;
  const size_t QN_BYTES   = (size_t)BATCH * QD * 2;   // 64 MB  bf16 q_hat
  const size_t KN_BYTES   = (size_t)NP    * QD * 2;   //  4 MB  bf16 k_hat
  const size_t DIST_BYTES = (size_t)BATCH * NP * 4;   // 64 MB  f32 dist
  const size_t SELW_BYTES = (size_t)BATCH * NSEL * 4; //  2 MB

  unsigned short* qn  = (unsigned short*)(ws);
  unsigned short* kn  = (unsigned short*)(ws + QN_BYTES);
  float* dist         = (float*)(ws + QN_BYTES + KN_BYTES);
  float* selw         = (float*)(ws + QN_BYTES + KN_BYTES + DIST_BYTES);
  int*   seli         = (int*)  (ws + QN_BYTES + KN_BYTES + DIST_BYTES + SELW_BYTES);

  float* outsel = out + (size_t)BATCH * N_CLOUD * OUT_DIM;  // selected_dist tail

  rownorm_bf16_kernel<<<BATCH, 256, 0, stream>>>(query, qn);
  rownorm_bf16_kernel<<<NP,    256, 0, stream>>>(keys,  kn);
  gemm_dist_kernel<<<dim3(NP / 256, BATCH / 64), 256, 0, stream>>>(qn, kn, dist);
  topk_kernel<<<BATCH, 256, 0, stream>>>(dist, selw, seli, outsel);
  gather_mlp_kernel<<<BATCH, 256, 0, stream>>>(prompts, selw, seli, w1, b1, w2, w2b, out);
}
